// TemporalAttention_34376918237507
// MI455X (gfx1250) — matrix operand order repaired and checked by reading, not compiled, not run on hardware
//
#include <hip/hip_runtime.h>
#include <hip/hip_bf16.h>

typedef __attribute__((ext_vector_type(16))) _Float16 v16h;
typedef __attribute__((ext_vector_type(8)))  float    v8f;

#define HIDDEN 1024
#define HEADS  16
#define HEAD   64
#define BATCH  2
#define SEQ    2048

// ---------- WMMA fragment helpers (wave32, 16x16x32 f16) ----------

// A-matrix 16x32: lane (m = lane&15, hi = lane>>4) holds K = {hi*8 .. +7, 16+hi*8 .. +7}
__device__ __forceinline__ v16h fragA_ld(const _Float16* base, int ld) {
    const int lane = threadIdx.x & 31;
    const int m = lane & 15, hi = lane >> 4;
    const _Float16* p = base + (size_t)m * ld + hi * 8;
    union { uint4 u[2]; v16h v; } f;
    f.u[0] = *reinterpret_cast<const uint4*>(p);
    f.u[1] = *reinterpret_cast<const uint4*>(p + 16);
    return f.v;
}

// B-matrix 32x16: lane (n = lane&15, hi = lane>>4) holds K = {hi*8 .. +7, 16+hi*8 .. +7}
// memory viewed as rows of the transposed B (i.e. [n][k] row-major with stride ld)
__device__ __forceinline__ v16h fragB_ld(const _Float16* base, int ld) {
    const int lane = threadIdx.x & 31;
    const int n = lane & 15, hi = lane >> 4;
    const _Float16* p = base + (size_t)n * ld + hi * 8;
    union { uint4 u[2]; v16h v; } f;
    f.u[0] = *reinterpret_cast<const uint4*>(p);
    f.u[1] = *reinterpret_cast<const uint4*>(p + 16);
    return f.v;
}

__device__ __forceinline__ v8f wmma_f16(v16h a, v16h b, v8f c) {
    return __builtin_amdgcn_wmma_f32_16x16x32_f16(false, a, false, b, (short)0, c, false, false);
}

// ---------- DPP16 row (16-lane) reductions: stay on the VALU pipe ----------
// row_ror:N rotates within each 16-lane row -> never crosses the half-wave.
#define DPP_ROR_F32(v, ctrl)                                                   \
    __builtin_bit_cast(float, __builtin_amdgcn_update_dpp(                     \
        0, __builtin_bit_cast(int, (v)), (ctrl), 0xf, 0xf, true))

__device__ __forceinline__ float row16_max(float v) {
    v = fmaxf(v, DPP_ROR_F32(v, 0x121));   // ror 1
    v = fmaxf(v, DPP_ROR_F32(v, 0x122));   // ror 2
    v = fmaxf(v, DPP_ROR_F32(v, 0x124));   // ror 4
    v = fmaxf(v, DPP_ROR_F32(v, 0x128));   // ror 8
    return v;
}

__device__ __forceinline__ float row16_sum(float v) {
    v += DPP_ROR_F32(v, 0x121);
    v += DPP_ROR_F32(v, 0x122);
    v += DPP_ROR_F32(v, 0x124);
    v += DPP_ROR_F32(v, 0x128);
    return v;
}

// ---------- Kernel 0: x + positional encoding -> f16 ----------
__global__ void __launch_bounds__(256)
k_add_pe(const float* __restrict__ x, _Float16* __restrict__ xh) {
    int i = blockIdx.x * 256 + threadIdx.x;          // 0 .. B*S*HIDDEN-1
    int hid = i & (HIDDEN - 1);
    int s   = (i >> 10) & (SEQ - 1);
    int pair = hid >> 1;
    float div = __expf((float)(2 * pair) * (-9.210340371976184f / (float)HIDDEN));
    float ang = (float)s * div;
    float pe  = (hid & 1) ? __cosf(ang) : __sinf(ang);
    xh[i] = (_Float16)(x[i] + pe);
}

// ---------- Kernel 1: weight f32 -> f16 ----------
__global__ void __launch_bounds__(256)
k_cvt_w(const float* __restrict__ W, _Float16* __restrict__ Wh) {
    int i = blockIdx.x * 256 + threadIdx.x;
    Wh[i] = (_Float16)W[i];
}

// ---------- Kernel 2: Y = X @ W^T + b  (32x64 tile / wave, double-buffered) ----------
// MODE 0: f16 out [b,h,s,d];  MODE 1: f16 out [b,h,d,s] (V);  MODE 2: f32 out row-major
template <int MODE>
__global__ void __launch_bounds__(256)
k_gemm(const _Float16* __restrict__ X, const _Float16* __restrict__ W,
       const float* __restrict__ bias, void* __restrict__ Yv) {
    const int lane = threadIdx.x & 31;
    const int wid  = (blockIdx.x * 256 + threadIdx.x) >> 5;   // 0..2047
    const int mt   = wid >> 4;                                // 0..127 (32 rows each)
    const int n0   = (wid & 15) * 64;                         // N strip base
    const int nl = lane & 15, hi = lane >> 4;

    const _Float16* xr0 = X + (size_t)(mt * 32) * HIDDEN;
    const _Float16* xr1 = xr0 + (size_t)16 * HIDDEN;
    const _Float16* wr  = W + (size_t)n0 * HIDDEN;

    v8f accA[4] = {{}, {}, {}, {}};
    v8f accB[4] = {{}, {}, {}, {}};

    // prologue: fragments for kk = 0
    v16h a0 = fragA_ld(xr0, HIDDEN);
    v16h a1 = fragA_ld(xr1, HIDDEN);
    v16h b[4];
    #pragma unroll
    for (int t = 0; t < 4; ++t) b[t] = fragB_ld(wr + (size_t)(t * 16) * HIDDEN, HIDDEN);

    // steady state: prefetch kk, compute kk-32
    for (int kk = 32; kk < HIDDEN; kk += 32) {
        v16h na0 = fragA_ld(xr0 + kk, HIDDEN);
        v16h na1 = fragA_ld(xr1 + kk, HIDDEN);
        v16h nb[4];
        #pragma unroll
        for (int t = 0; t < 4; ++t) nb[t] = fragB_ld(wr + (size_t)(t * 16) * HIDDEN + kk, HIDDEN);
        #pragma unroll
        for (int t = 0; t < 4; ++t) {
            accA[t] = wmma_f16(a0, b[t], accA[t]);
            accB[t] = wmma_f16(a1, b[t], accB[t]);
        }
        a0 = na0; a1 = na1;
        #pragma unroll
        for (int t = 0; t < 4; ++t) b[t] = nb[t];
    }
    // tail
    #pragma unroll
    for (int t = 0; t < 4; ++t) {
        accA[t] = wmma_f16(a0, b[t], accA[t]);
        accB[t] = wmma_f16(a1, b[t], accB[t]);
    }

    // epilogue
    #pragma unroll
    for (int t = 0; t < 4; ++t) {
        int col = n0 + t * 16 + nl;
        float bvv = bias[col];
        int h = col >> 6, d = col & 63;
        #pragma unroll
        for (int half = 0; half < 2; ++half) {
            v8f acc = half ? accB[t] : accA[t];
            #pragma unroll
            for (int r = 0; r < 8; ++r) {
                int row = mt * 32 + half * 16 + r + 8 * hi;   // 0..4095
                float v = acc[r] + bvv;
                if (MODE == 0) {
                    int bb = row >> 11, ss = row & (SEQ - 1);
                    ((_Float16*)Yv)[(((size_t)(bb * HEADS + h)) * SEQ + ss) * HEAD + d] = (_Float16)v;
                } else if (MODE == 1) {
                    int bb = row >> 11, ss = row & (SEQ - 1);
                    ((_Float16*)Yv)[(((size_t)(bb * HEADS + h)) * HEAD + d) * SEQ + ss] = (_Float16)v;
                } else {
                    ((float*)Yv)[(size_t)row * HIDDEN + col] = v;
                }
            }
        }
    }
}

// ---------- Kernel 3: flash attention ----------
// wave = (b,h, 16-query block); stream keys in 32-key blocks with online softmax.
// V B-fragments load at the top of each block so they are in flight during the
// score WMMAs and the softmax VALU work. Row reductions are DPP16 rotations
// (VALU, co-executes with WMMA); the DS pipe carries only the P staging.
__global__ void __launch_bounds__(256)
k_attn(const _Float16* __restrict__ Q, const _Float16* __restrict__ K,
       const _Float16* __restrict__ Vt, _Float16* __restrict__ ctx) {
    __shared__ _Float16 plds[8][16 * 40];         // per-wave P staging (40-half row stride)
    const int lane = threadIdx.x & 31;
    const int wib  = threadIdx.x >> 5;
    const int wid  = blockIdx.x * 8 + wib;        // 0..4095
    const int bh   = wid >> 7;                    // 0..31  (b*16+h)
    const int qb   = wid & 127;                   // query block
    const int nl = lane & 15, hi = lane >> 4;

    const _Float16* Qbase = Q  + ((size_t)bh * SEQ + qb * 16) * HEAD;
    const _Float16* Kbase = K  + (size_t)bh * SEQ * HEAD;
    const _Float16* Vbase = Vt + (size_t)bh * HEAD * SEQ;
    _Float16* pl = plds[wib];

    v16h qa0 = fragA_ld(Qbase,      HEAD);        // d 0..31
    v16h qa1 = fragA_ld(Qbase + 32, HEAD);        // d 32..63

    v8f acc[4] = {{}, {}, {}, {}};
    float mrow[8], lrow[8];
    #pragma unroll
    for (int r = 0; r < 8; ++r) { mrow[r] = -1e30f; lrow[r] = 0.0f; }

    for (int kb = 0; kb < SEQ; kb += 32) {
        // ---- issue all global loads for this block up front ----
        v16h kf0 = fragB_ld(Kbase + (size_t)kb        * HEAD,      HEAD);
        v16h kf1 = fragB_ld(Kbase + (size_t)kb        * HEAD + 32, HEAD);
        v16h kf2 = fragB_ld(Kbase + (size_t)(kb + 16) * HEAD,      HEAD);
        v16h kf3 = fragB_ld(Kbase + (size_t)(kb + 16) * HEAD + 32, HEAD);
        v16h vf[4];
        #pragma unroll
        for (int t = 0; t < 4; ++t)
            vf[t] = fragB_ld(Vbase + (size_t)(t * 16) * SEQ + kb, SEQ);

        // ---- scores: 16q x 32k tile ----
        v8f c0 = {}, c1 = {};
        c0 = wmma_f16(qa0, kf0, c0);
        c0 = wmma_f16(qa1, kf1, c0);
        c1 = wmma_f16(qa0, kf2, c1);
        c1 = wmma_f16(qa1, kf3, c1);

        // ---- online softmax (rows live across 16 lanes of each half-wave) ----
        #pragma unroll
        for (int r = 0; r < 8; ++r) {
            float s0 = c0[r] * 0.125f;            // 1/sqrt(64)
            float s1 = c1[r] * 0.125f;
            float mx = row16_max(fmaxf(s0, s1));
            float mnew  = fmaxf(mrow[r], mx);
            float alpha = __expf(mrow[r] - mnew);
            float p0 = __expf(s0 - mnew);
            float p1 = __expf(s1 - mnew);
            float ps = row16_sum(p0 + p1);
            lrow[r] = lrow[r] * alpha + ps;
            mrow[r] = mnew;
            #pragma unroll
            for (int t = 0; t < 4; ++t) acc[t][r] *= alpha;
            int m = r + 8 * hi;
            pl[m * 40 + nl]      = (_Float16)p0;  // keys kb..kb+15
            pl[m * 40 + 16 + nl] = (_Float16)p1;  // keys kb+16..kb+31
        }

        // ---- reload P as an A-fragment from LDS (same-wave DS ops are in-order) ----
        union { uint4 u[2]; v16h v; } pf;
        {
            const _Float16* pp = pl + (size_t)nl * 40 + hi * 8;
            pf.u[0] = *reinterpret_cast<const uint4*>(pp);
            pf.u[1] = *reinterpret_cast<const uint4*>(pp + 16);
        }

        // ---- ctx += P @ V  (V transposed: Vt[d][s], contiguous along keys) ----
        #pragma unroll
        for (int t = 0; t < 4; ++t) acc[t] = wmma_f16(pf.v, vf[t], acc[t]);
    }

    // ---- epilogue: normalize and store ctx as [b, s, h*64+d] f16 ----
    int b = bh >> 4, h = bh & 15;
    #pragma unroll
    for (int r = 0; r < 8; ++r) {
        float inv = 1.0f / lrow[r];
        int q = qb * 16 + r + 8 * hi;
        _Float16* dst = ctx + ((size_t)(b * SEQ + q)) * HIDDEN + h * HEAD + nl;
        dst[0]  = (_Float16)(acc[0][r] * inv);
        dst[16] = (_Float16)(acc[1][r] * inv);
        dst[32] = (_Float16)(acc[2][r] * inv);
        dst[48] = (_Float16)(acc[3][r] * inv);
    }
}

// ---------- host launcher ----------
extern "C" void kernel_launch(void* const* d_in, const int* in_sizes, int n_in,
                              void* d_out, int out_size, void* d_ws, size_t ws_size,
                              hipStream_t stream) {
    const float* x  = (const float*)d_in[0];
    const float* Wq = (const float*)d_in[1];
    const float* bq = (const float*)d_in[2];
    const float* Wk = (const float*)d_in[3];
    const float* bk = (const float*)d_in[4];
    const float* Wv = (const float*)d_in[5];
    const float* bv = (const float*)d_in[6];
    const float* Wo = (const float*)d_in[7];
    const float* bo = (const float*)d_in[8];

    char* ws = (char*)d_ws;
    _Float16* xh   = (_Float16*)(ws);                          // 8 MB
    _Float16* wqh  = (_Float16*)(ws + ((size_t) 8 << 20));     // 2 MB
    _Float16* wkh  = (_Float16*)(ws + ((size_t)10 << 20));     // 2 MB
    _Float16* wvh  = (_Float16*)(ws + ((size_t)12 << 20));     // 2 MB
    _Float16* woh  = (_Float16*)(ws + ((size_t)14 << 20));     // 2 MB
    _Float16* Qh   = (_Float16*)(ws + ((size_t)16 << 20));     // 8 MB [b,h,s,d]
    _Float16* Kh   = (_Float16*)(ws + ((size_t)24 << 20));     // 8 MB [b,h,s,d]
    _Float16* Vth  = (_Float16*)(ws + ((size_t)32 << 20));     // 8 MB [b,h,d,s]
    _Float16* ctxh = (_Float16*)(ws + ((size_t)40 << 20));     // 8 MB [b,s,hid]

    k_add_pe<<<16384, 256, 0, stream>>>(x, xh);
    k_cvt_w <<< 4096, 256, 0, stream>>>(Wq, wqh);
    k_cvt_w <<< 4096, 256, 0, stream>>>(Wk, wkh);
    k_cvt_w <<< 4096, 256, 0, stream>>>(Wv, wvh);
    k_cvt_w <<< 4096, 256, 0, stream>>>(Wo, woh);

    k_gemm<0><<<256, 256, 0, stream>>>(xh, wqh, bq, (void*)Qh);
    k_gemm<0><<<256, 256, 0, stream>>>(xh, wkh, bk, (void*)Kh);
    k_gemm<1><<<256, 256, 0, stream>>>(xh, wvh, bv, (void*)Vth);

    k_attn<<<512, 256, 0, stream>>>(Qh, Kh, Vth, ctxh);

    k_gemm<2><<<256, 256, 0, stream>>>(ctxh, woh, bo, d_out);
}
